// MultiHeadTemporalAttention_62732292325771
// MI455X (gfx1250) — compile-verified, matrix operand order checked by reading, not run on hardware
//
#include <hip/hip_runtime.h>
#include <hip/hip_bf16.h>

// ---------------- problem constants (fixed by reference) ----------------
constexpr int Bc  = 64;
constexpr int Sc  = 200;
constexpr int Hc  = 1024;
constexpr int NHc = 16;
constexpr int HDc = 64;
constexpr int FFc = 4096;
constexpr int Mrows = Bc * Sc;          // 12800
constexpr int PEROWS = 399;             // 2*MAX_LEN-1
constexpr int PEPAD  = 416;             // padded to 26 tiles of 16

typedef unsigned int  u32;
typedef unsigned short u16;
typedef __attribute__((ext_vector_type(4)))  u32    v4u;
typedef __attribute__((ext_vector_type(8)))  u32    v8u;
typedef __attribute__((ext_vector_type(8)))  float  v8f;
typedef __attribute__((ext_vector_type(16))) __bf16 v16bf;

static __device__ __forceinline__ int imin(int a, int b) { return a < b ? a : b; }

static __device__ __forceinline__ u16 f2bf(float f) {
  u32 u = __float_as_uint(f);
  u += 0x7FFFu + ((u >> 16) & 1u);      // round-to-nearest-even
  return (u16)(u >> 16);
}

static __device__ __forceinline__ v8f wmma_bf16(v8u a, v8u b, v8f c) {
  return __builtin_amdgcn_wmma_f32_16x16x32_bf16(
      false, __builtin_bit_cast(v16bf, a),
      false, __builtin_bit_cast(v16bf, b),
      (short)0, c, false, false);
}

// A/B fragment loader for a 32-wide bf16 K-chunk starting at p (16B aligned).
// Lane half 'half' (lane>>4): elements half*8+{0..7} and 16+half*8+{0..7}.
static __device__ __forceinline__ v8u load_row_frag(const u16* p, int half) {
  v4u a = *reinterpret_cast<const v4u*>(p + half * 8);
  v4u b = *reinterpret_cast<const v4u*>(p + 16 + half * 8);
  v8u r;
  r[0] = a[0]; r[1] = a[1]; r[2] = a[2]; r[3] = a[3];
  r[4] = b[0]; r[5] = b[1]; r[6] = b[2]; r[7] = b[3];
  return r;
}

// ---------------- weight cast/transpose: f32 [K,N] -> bf16 [N,K] ----------------
__global__ __launch_bounds__(256) void transpose_cast(const float* __restrict__ in,
                                                      u16* __restrict__ out,
                                                      int K, int N) {
  long total = (long)K * N;
  long idx = (long)blockIdx.x * blockDim.x + threadIdx.x;
  if (idx >= total) return;
  int k = (int)(idx / N);
  int n = (int)(idx % N);
  out[(size_t)n * K + k] = f2bf(in[idx]);
}

// cast pos_embed [399,64] -> bf16 [416,64] with zero padding
__global__ __launch_bounds__(256) void cast_pe(const float* __restrict__ in,
                                               u16* __restrict__ out) {
  int idx = blockIdx.x * blockDim.x + threadIdx.x;
  if (idx >= PEPAD * HDc) return;
  out[idx] = (idx < PEROWS * HDc) ? f2bf(in[idx]) : (u16)0;
}

// ---------------- fused LayerNorm (+ optional time embedding) -> bf16 ----------------
__global__ __launch_bounds__(256) void ln_embed(const float* __restrict__ x,
                                                const float* __restrict__ tvec,
                                                const float* __restrict__ wt,
                                                const float* __restrict__ bt,
                                                const float* __restrict__ g,
                                                const float* __restrict__ be,
                                                u16* __restrict__ out,
                                                int useTime) {
  __shared__ float red[256];
  int row = blockIdx.x;                 // B*S rows
  int tid = threadIdx.x;
  const float* xr = x + (size_t)row * Hc;

  float s = 0.f;
  for (int i = tid; i < Hc; i += 256) s += xr[i];
  red[tid] = s; __syncthreads();
  for (int st = 128; st > 0; st >>= 1) { if (tid < st) red[tid] += red[tid + st]; __syncthreads(); }
  float mean = red[0] * (1.0f / Hc);
  __syncthreads();

  float s2 = 0.f;
  for (int i = tid; i < Hc; i += 256) { float d = xr[i] - mean; s2 += d * d; }
  red[tid] = s2; __syncthreads();
  for (int st = 128; st > 0; st >>= 1) { if (tid < st) red[tid] += red[tid + st]; __syncthreads(); }
  float rstd = rsqrtf(red[0] * (1.0f / Hc) + 1e-5f);

  float ts = useTime ? tvec[row] : 0.f;
  for (int i = tid; i < Hc; i += 256) {
    float v = (xr[i] - mean) * rstd * g[i] + be[i];
    if (useTime) v += ts * wt[i] + bt[i];
    out[(size_t)row * Hc + i] = f2bf(v);
  }
}

// ---------------- WMMA bf16 GEMM: C = epilogue(A[M,K] @ Wt[N,K]^T + bias) ----------------
// Wave tile 32x64 (2 A-frags x 4 B-frags -> 8 WMMA per 32-K chunk).
// Block: 8 waves as 4(M) x 2(N) -> 128x128 block tile.
// MODE 0: store bf16           MODE 1: +resid(f32), store f32
// MODE 2: GELU, store bf16     MODE 3: +resid(f32), store f32
template <int MODE>
__global__ __launch_bounds__(256) void gemm_bf16(const u16* __restrict__ A,
                                                 const u16* __restrict__ Wt,
                                                 const float* __restrict__ bias,
                                                 const float* __restrict__ resid,
                                                 void* __restrict__ outv,
                                                 int K, int N) {
  int lane = threadIdx.x & 31;
  int w    = threadIdx.x >> 5;
  int wm   = w & 3, wn = w >> 2;        // 4x2 waves
  int m0 = blockIdx.x * 128 + wm * 32;
  int n0 = blockIdx.y * 128 + wn * 64;
  int half = lane >> 4, t = lane & 15;

  const u16* ar0 = A  + (size_t)(m0 + t) * K;
  const u16* ar1 = A  + (size_t)(m0 + 16 + t) * K;
  const u16* br0 = Wt + (size_t)(n0 + t) * K;
  const u16* br1 = Wt + (size_t)(n0 + 16 + t) * K;
  const u16* br2 = Wt + (size_t)(n0 + 32 + t) * K;
  const u16* br3 = Wt + (size_t)(n0 + 48 + t) * K;

  v8f c00 = {}, c01 = {}, c02 = {}, c03 = {};
  v8f c10 = {}, c11 = {}, c12 = {}, c13 = {};

  for (int kb = 0; kb < K; kb += 32) {
    if ((kb & 127) == 0 && kb + 256 < K) {   // stream-ahead prefetch (512B/row)
      __builtin_prefetch(ar0 + kb + 256, 0, 1);
      __builtin_prefetch(ar1 + kb + 256, 0, 1);
      __builtin_prefetch(br0 + kb + 256, 0, 1);
      __builtin_prefetch(br1 + kb + 256, 0, 1);
      __builtin_prefetch(br2 + kb + 256, 0, 1);
      __builtin_prefetch(br3 + kb + 256, 0, 1);
    }
    v8u a0 = load_row_frag(ar0 + kb, half);
    v8u a1 = load_row_frag(ar1 + kb, half);
    v8u b0 = load_row_frag(br0 + kb, half);
    v8u b1 = load_row_frag(br1 + kb, half);
    v8u b2 = load_row_frag(br2 + kb, half);
    v8u b3 = load_row_frag(br3 + kb, half);
    c00 = wmma_bf16(a0, b0, c00);
    c01 = wmma_bf16(a0, b1, c01);
    c02 = wmma_bf16(a0, b2, c02);
    c03 = wmma_bf16(a0, b3, c03);
    c10 = wmma_bf16(a1, b0, c10);
    c11 = wmma_bf16(a1, b1, c11);
    c12 = wmma_bf16(a1, b2, c12);
    c13 = wmma_bf16(a1, b3, c13);
  }

  v8f cs[2][4] = {{c00, c01, c02, c03}, {c10, c11, c12, c13}};
#pragma unroll
  for (int i = 0; i < 2; ++i) {
#pragma unroll
    for (int j = 0; j < 4; ++j) {
      int n = n0 + j * 16 + t;
      float bn = bias[n];
#pragma unroll
      for (int r = 0; r < 8; ++r) {
        int m = m0 + i * 16 + half * 8 + r;
        float v = cs[i][j][r] + bn;
        if (MODE == 2) v = 0.5f * v * (1.0f + erff(v * 0.7071067811865476f));
        size_t idx = (size_t)m * N + n;
        if (MODE == 0 || MODE == 2) {
          ((u16*)outv)[idx] = f2bf(v);
        } else {
          ((float*)outv)[idx] = v + resid[idx];
        }
      }
    }
  }
}

// ---------------- attention: one wave per (b, head, 16-query tile) ----------------
static __device__ __forceinline__ v8u load_v_frag(const u16* __restrict__ vb,
                                                  int rowBase, int colIdx,
                                                  int kchunk, int half) {
  v8u r;
#pragma unroll
  for (int j = 0; j < 8; ++j) {
    int koff = (j < 4) ? (half * 8 + 2 * j) : (16 + half * 8 + 2 * (j - 4));
    int k0e  = kchunk + koff;
    int r0 = imin(k0e,     Sc - 1);
    int r1 = imin(k0e + 1, Sc - 1);
    u32 lo = vb[(size_t)(rowBase + r0) * Hc + colIdx];
    u32 hi = vb[(size_t)(rowBase + r1) * Hc + colIdx];
    r[j] = lo | (hi << 16);
  }
  return r;
}

__global__ __launch_bounds__(32) void attn_kernel(const u16* __restrict__ qb,
                                                  const u16* __restrict__ kbuf,
                                                  const u16* __restrict__ vbuf,
                                                  const u16* __restrict__ pe,
                                                  u16* __restrict__ ctx) {
  __shared__ __attribute__((aligned(16))) float sc[16][224];
  __shared__ __attribute__((aligned(16))) float tb[16][32];
  __shared__ __attribute__((aligned(16))) u16   P[16][224];
  __shared__ float invs[16];

  int qt = blockIdx.x, hh = blockIdx.y, b = blockIdx.z;
  int q0 = qt * 16;
  int lane = threadIdx.x;
  int half = lane >> 4, t = lane & 15;
  int nkt = qt + 1;
  int ncol = nkt * 16;
  int nchunks = (ncol + 31) / 32;
  int npad = nchunks * 32;
  int rowBase = b * Sc;

  // Q fragments (A-matrix): reused for scores and rel-bias
  int qrow = imin(q0 + t, Sc - 1);
  const u16* qp = qb + ((size_t)(rowBase + qrow) * Hc + hh * HDc);
  v8u aQ0 = load_row_frag(qp, half);
  v8u aQ1 = load_row_frag(qp + 32, half);

  const float scale = 0.125f;           // 1/sqrt(64)

  for (int kt = 0; kt < nkt; ++kt) {
    int k0 = kt * 16;
    int krow = imin(k0 + t, Sc - 1);
    const u16* kp = kbuf + ((size_t)(rowBase + krow) * Hc + hh * HDc);
    v8u bK0 = load_row_frag(kp, half);
    v8u bK1 = load_row_frag(kp + 32, half);
    v8f c = {};
    c = wmma_bf16(aQ0, bK0, c);
    c = wmma_bf16(aQ1, bK1, c);

    // relative-position bias window: T = Qtile @ PE[base .. base+32)^T
    int base = q0 - k0 + 184;           // in [184, 376]; +31 < 416 (padded)
#pragma unroll
    for (int tt = 0; tt < 2; ++tt) {
      const u16* pp = pe + (size_t)(base + tt * 16 + t) * HDc;
      v8u bP0 = load_row_frag(pp, half);
      v8u bP1 = load_row_frag(pp + 32, half);
      v8f tv = {};
      tv = wmma_bf16(aQ0, bP0, tv);
      tv = wmma_bf16(aQ1, bP1, tv);
#pragma unroll
      for (int r = 0; r < 8; ++r) tb[half * 8 + r][tt * 16 + t] = tv[r];
    }
    __syncthreads();

#pragma unroll
    for (int r = 0; r < 8; ++r) {
      int m = half * 8 + r;
      int q = q0 + m, k = k0 + t;
      float v = c[r] * scale + tb[m][m - t + 15];   // rel = q-k+199 gather
      if (k > q || k >= Sc || q >= Sc) v = -1e9f;
      sc[m][k0 + t] = v;
    }
    __syncthreads();
  }

  // softmax per query row (unnormalized exp in bf16; normalize via invs at end)
  if (lane < 16) {
    int m = lane;
    float mx = -3.4e38f;
    for (int j = 0; j < ncol; ++j) mx = fmaxf(mx, sc[m][j]);
    float sum = 0.f;
    for (int j = 0; j < ncol; ++j) {
      float e = __expf(sc[m][j] - mx);
      sum += e;
      P[m][j] = f2bf(e);
    }
    for (int j = ncol; j < npad; ++j) P[m][j] = 0;
    invs[m] = 1.0f / sum;
  }
  __syncthreads();

  // ctx = P @ V   (16 x 64 output, 4 d-tiles)
  v8f acc[4] = {};
  for (int cc = 0; cc < nchunks; ++cc) {
    const u16* pr = &P[t][cc * 32];
    v8u aP = load_row_frag(pr, half);
#pragma unroll
    for (int dt = 0; dt < 4; ++dt) {
      v8u bV = load_v_frag(vbuf, rowBase, hh * HDc + dt * 16 + t, cc * 32, half);
      acc[dt] = wmma_bf16(aP, bV, acc[dt]);
    }
  }

#pragma unroll
  for (int dt = 0; dt < 4; ++dt) {
    int d = hh * HDc + dt * 16 + t;
#pragma unroll
    for (int r = 0; r < 8; ++r) {
      int m = half * 8 + r;
      int q = q0 + m;
      if (q < Sc) ctx[(size_t)(rowBase + q) * Hc + d] = f2bf(acc[dt][r] * invs[m]);
    }
  }
}

// ---------------- host orchestration ----------------
extern "C" void kernel_launch(void* const* d_in, const int* in_sizes, int n_in,
                              void* d_out, int out_size, void* d_ws, size_t ws_size,
                              hipStream_t stream) {
  const float* x    = (const float*)d_in[0];
  const float* timv = (const float*)d_in[1];
  const float* wq   = (const float*)d_in[2];
  const float* bq   = (const float*)d_in[3];
  const float* wk   = (const float*)d_in[4];
  const float* bk   = (const float*)d_in[5];
  const float* wv   = (const float*)d_in[6];
  const float* bv   = (const float*)d_in[7];
  const float* wo   = (const float*)d_in[8];
  const float* bo   = (const float*)d_in[9];
  const float* wt   = (const float*)d_in[10];
  const float* bt   = (const float*)d_in[11];
  const float* pos  = (const float*)d_in[12];
  const float* g1   = (const float*)d_in[13];
  const float* be1  = (const float*)d_in[14];
  const float* g2   = (const float*)d_in[15];
  const float* be2  = (const float*)d_in[16];
  const float* w1   = (const float*)d_in[17];
  const float* bf1  = (const float*)d_in[18];
  const float* w2   = (const float*)d_in[19];
  const float* bf2  = (const float*)d_in[20];

  char* p = (char*)d_ws;
  auto alloc = [&](size_t bytes) -> void* {
    void* r = (void*)p;
    p += (bytes + 255) & ~(size_t)255;
    return r;
  };

  u16* h    = (u16*)alloc((size_t)Mrows * Hc * 2);
  u16* qb   = (u16*)alloc((size_t)Mrows * Hc * 2);
  u16* kb   = (u16*)alloc((size_t)Mrows * Hc * 2);
  u16* vb   = (u16*)alloc((size_t)Mrows * Hc * 2);
  float* o1 = (float*)alloc((size_t)Mrows * Hc * 4);
  u16* ffh  = (u16*)alloc((size_t)Mrows * FFc * 2);
  u16* WqT  = (u16*)alloc((size_t)Hc * Hc * 2);
  u16* WkT  = (u16*)alloc((size_t)Hc * Hc * 2);
  u16* WvT  = (u16*)alloc((size_t)Hc * Hc * 2);
  u16* WoT  = (u16*)alloc((size_t)Hc * Hc * 2);
  u16* W1T  = (u16*)alloc((size_t)Hc * FFc * 2);
  u16* W2T  = (u16*)alloc((size_t)FFc * Hc * 2);
  u16* PE   = (u16*)alloc((size_t)PEPAD * HDc * 2);
  u16* ctx  = h;    // alias: h dead after QKV GEMMs
  u16* h2   = qb;   // alias: q dead after attention

  // 1) weight casts/transposes
  {
    long t1 = (long)Hc * Hc;
    int blk1 = (int)((t1 + 255) / 256);
    transpose_cast<<<blk1, 256, 0, stream>>>(wq, WqT, Hc, Hc);
    transpose_cast<<<blk1, 256, 0, stream>>>(wk, WkT, Hc, Hc);
    transpose_cast<<<blk1, 256, 0, stream>>>(wv, WvT, Hc, Hc);
    transpose_cast<<<blk1, 256, 0, stream>>>(wo, WoT, Hc, Hc);
    long t2 = (long)Hc * FFc;
    int blk2 = (int)((t2 + 255) / 256);
    transpose_cast<<<blk2, 256, 0, stream>>>(w1, W1T, Hc, FFc);
    transpose_cast<<<blk2, 256, 0, stream>>>(w2, W2T, FFc, Hc);
    cast_pe<<<(PEPAD * HDc + 255) / 256, 256, 0, stream>>>(pos, PE);
  }

  // 2) h = LN1(x) + time*wt + bt   (bf16)
  ln_embed<<<Mrows, 256, 0, stream>>>(x, timv, wt, bt, g1, be1, h, 1);

  // 3) Q,K,V projections (bf16 out)
  dim3 gP(Mrows / 128, Hc / 128);
  gemm_bf16<0><<<gP, 256, 0, stream>>>(h, WqT, bq, nullptr, qb, Hc, Hc);
  gemm_bf16<0><<<gP, 256, 0, stream>>>(h, WkT, bk, nullptr, kb, Hc, Hc);
  gemm_bf16<0><<<gP, 256, 0, stream>>>(h, WvT, bv, nullptr, vb, Hc, Hc);

  // 4) attention (scores + rel-bias + causal softmax + P@V)
  dim3 gA((Sc + 15) / 16, NHc, Bc);
  attn_kernel<<<gA, 32, 0, stream>>>(qb, kb, vb, PE, ctx);

  // 5) out1 = ctx @ Wo + bo + x (f32)
  gemm_bf16<1><<<gP, 256, 0, stream>>>(ctx, WoT, bo, x, o1, Hc, Hc);

  // 6) h2 = LN2(out1) (bf16)
  ln_embed<<<Mrows, 256, 0, stream>>>(o1, nullptr, nullptr, nullptr, g2, be2, h2, 0);

  // 7) ffh = gelu(h2 @ W1 + bf1) (bf16)
  dim3 gF1(Mrows / 128, FFc / 128);
  gemm_bf16<2><<<gF1, 256, 0, stream>>>(h2, W1T, bf1, nullptr, ffh, Hc, FFc);

  // 8) out = ffh @ W2 + bf2 + out1 (f32)
  gemm_bf16<3><<<gP, 256, 0, stream>>>(ffh, W2T, bf2, o1, d_out, FFc, Hc);
}
